// MultiScaleFus_82248623718708
// MI455X (gfx1250) — matrix operand order repaired and checked by reading, not compiled, not run on hardware
//
#include <hip/hip_runtime.h>
#include <hip/hip_bf16.h>

typedef __attribute__((ext_vector_type(16))) __bf16 v16bf;
typedef __attribute__((ext_vector_type(8)))  float  v8f;

#define BATCH 32
#define NSC 4
#define LOOK 2048
#define DIM 128

// ---------------------------------------------------------------------------
// Kernel A: per (b,n) fold Q projection into K-score vector.
//   qhat = Wq @ q_in + bq
//   kqs[b,n,:] = scale * (Wk^T @ qhat);   cks[b,n] = scale * <qhat, bk>
// ---------------------------------------------------------------------------
__global__ __launch_bounds__(128) void ms_prep(const float* __restrict__ Qin,
                                               const float* __restrict__ Wq,
                                               const float* __restrict__ bq,
                                               const float* __restrict__ Wk,
                                               const float* __restrict__ bk,
                                               float* __restrict__ kqs,
                                               float* __restrict__ cks) {
    __shared__ float qh[DIM];
    __shared__ float pr[DIM];
    const int bn = blockIdx.x;      // b*NSC + n, 0..127
    const int j  = threadIdx.x;     // 0..127
    const float* q   = Qin + bn * DIM;
    const float* wqr = Wq + j * DIM;
    float acc = bq[j];
    #pragma unroll 4
    for (int d = 0; d < DIM; ++d) acc += wqr[d] * q[d];
    qh[j] = acc;
    pr[j] = acc * bk[j];
    __syncthreads();
    const float scale = 0.08838834764831845f;   // 1/sqrt(128)
    float kq = 0.f;                              // column j of Wk^T @ qhat
    #pragma unroll 4
    for (int jj = 0; jj < DIM; ++jj) kq += Wk[jj * DIM + j] * qh[jj];
    kqs[bn * DIM + j] = kq * scale;
    if (j == 0) {
        float s = 0.f;
        for (int jj = 0; jj < DIM; ++jj) s += pr[jj];
        cks[bn] = s * scale;
    }
}

// ---------------------------------------------------------------------------
// Kernel B: streaming pass. One wave per (b,l): 4 score dot-products against
// K_inputs, softmax over the 4 scales, weighted sum of V_inputs rows.
// Pure bandwidth: reads 268 MB once, writes 33.5 MB vbar.
// ---------------------------------------------------------------------------
__global__ __launch_bounds__(256) void ms_attn(const float* __restrict__ Kin,
                                               const float* __restrict__ Vin,
                                               const float* __restrict__ kqs,
                                               const float* __restrict__ cks,
                                               float* __restrict__ vbar) {
    const int  lane = threadIdx.x & 31;
    const long idx  = (long)blockIdx.x * 8 + (threadIdx.x >> 5);  // 0..65535
    const int  b    = (int)(idx >> 11);
    const int  l    = (int)(idx & (LOOK - 1));

    const float4* K4  = (const float4*)Kin;
    const float4* V4  = (const float4*)Vin;
    const float4* KQ4 = (const float4*)kqs;

    float s[NSC];
    #pragma unroll
    for (int n = 0; n < NSC; ++n) {
        const long base4 = ((long)(b * NSC + n) * LOOK + l) * (DIM / 4) + lane;
        const float4 kv = K4[base4];
        const float4 kq = KQ4[(b * NSC + n) * (DIM / 4) + lane];
        float p = kv.x * kq.x + kv.y * kq.y + kv.z * kq.z + kv.w * kq.w;
        #pragma unroll
        for (int m = 16; m >= 1; m >>= 1) p += __shfl_xor(p, m, 32);
        s[n] = p + cks[b * NSC + n];
    }
    const float mx = fmaxf(fmaxf(s[0], s[1]), fmaxf(s[2], s[3]));
    float e[NSC], sum = 0.f;
    #pragma unroll
    for (int n = 0; n < NSC; ++n) { e[n] = __expf(s[n] - mx); sum += e[n]; }
    const float inv = 1.f / sum;

    float4 acc = make_float4(0.f, 0.f, 0.f, 0.f);
    #pragma unroll
    for (int n = 0; n < NSC; ++n) {
        const long base4 = ((long)(b * NSC + n) * LOOK + l) * (DIM / 4) + lane;
        const float4 vv = V4[base4];
        const float a = e[n] * inv;
        acc.x += a * vv.x; acc.y += a * vv.y; acc.z += a * vv.z; acc.w += a * vv.w;
    }
    ((float4*)vbar)[idx * (DIM / 4) + lane] = acc;
}

// ---------------------------------------------------------------------------
// Kernel C: out = vbar @ Wv^T + bv via v_wmma_f32_16x16x32_bf16 with hi/lo
// bf16 split (3 WMMAs ~ f32 accuracy). 256 threads = 8 waves; block covers
// 128 rows x 128 cols; each wave does one 16-row stripe, looping 8 N-tiles
// and 4 K-chunks of 32.
// ---------------------------------------------------------------------------
__device__ __forceinline__ void split1(float x, v16bf& h, v16bf& l, int i) {
    __bf16 t = (__bf16)x;
    h[i] = t;
    l[i] = (__bf16)(x - (float)t);
}
__device__ __forceinline__ void split4(float4 f, v16bf& h, v16bf& l, int off) {
    split1(f.x, h, l, off + 0);
    split1(f.y, h, l, off + 1);
    split1(f.z, h, l, off + 2);
    split1(f.w, h, l, off + 3);
}

__global__ __launch_bounds__(256) void ms_proj(const float* __restrict__ vbar,
                                               const float* __restrict__ Wv,
                                               const float* __restrict__ bv,
                                               float* __restrict__ out) {
    const int w    = threadIdx.x >> 5;
    const int lane = threadIdx.x & 31;
    const int hf   = lane >> 4;     // which 16-lane half
    const int n16  = lane & 15;

    // --- A fragments (vbar rows), ISA 16-bit A 16x32 layout ---
    // lane half 0: K = kc+0..7 then kc+16..23 ; half 1: K = kc+8..15 then kc+24..31
    const long mrow = (long)blockIdx.x * 128 + w * 16 + n16;
    const float* arow = vbar + mrow * DIM;
    v16bf ah[4], al[4];
    #pragma unroll
    for (int kc4 = 0; kc4 < 4; ++kc4) {
        const float* p = arow + kc4 * 32 + 8 * hf;
        split4(*(const float4*)(p + 0), ah[kc4], al[kc4], 0);
        split4(*(const float4*)(p + 4), ah[kc4], al[kc4], 4);
        const float* q = arow + kc4 * 32 + 16 + 8 * hf;
        split4(*(const float4*)(q + 0), ah[kc4], al[kc4], 8);
        split4(*(const float4*)(q + 4), ah[kc4], al[kc4], 12);
    }

    const long obase = (long)blockIdx.x * 128 + w * 16;
    #pragma unroll
    for (int nj = 0; nj < 8; ++nj) {
        const int j0 = nj * 16;
        // bias: every element of this lane's v8f shares output column j0+n16
        const float bias = bv[j0 + n16];
        v8f acc;
        #pragma unroll
        for (int r = 0; r < 8; ++r) acc[r] = bias;

        #pragma unroll
        for (int kc4 = 0; kc4 < 4; ++kc4) {
            // B[k,n] = Wv[j0+n, kc+k]: lane reads K = kc+8*hf+0..7 then kc+16+8*hf+0..7 of Wv row
            const float* brow = Wv + (long)(j0 + n16) * DIM + kc4 * 32 + 8 * hf;
            v16bf bh, bl;
            split4(*(const float4*)(brow + 0),  bh, bl, 0);
            split4(*(const float4*)(brow + 4),  bh, bl, 4);
            split4(*(const float4*)(brow + 16), bh, bl, 8);
            split4(*(const float4*)(brow + 20), bh, bl, 12);
            acc = __builtin_amdgcn_wmma_f32_16x16x32_bf16(false, ah[kc4], false, bh,
                                                          (short)0, acc, false, false);
            acc = __builtin_amdgcn_wmma_f32_16x16x32_bf16(false, ah[kc4], false, bl,
                                                          (short)0, acc, false, false);
            acc = __builtin_amdgcn_wmma_f32_16x16x32_bf16(false, al[kc4], false, bh,
                                                          (short)0, acc, false, false);
        }
        // C/D layout: v8f element r -> row r + 8*hf, column n16
        #pragma unroll
        for (int r = 0; r < 8; ++r)
            out[(obase + r + 8 * hf) * DIM + j0 + n16] = acc[r];
    }
}

// ---------------------------------------------------------------------------
extern "C" void kernel_launch(void* const* d_in, const int* in_sizes, int n_in,
                              void* d_out, int out_size, void* d_ws, size_t ws_size,
                              hipStream_t stream) {
    const float* Qin = (const float*)d_in[0];
    const float* Kin = (const float*)d_in[1];
    const float* Vin = (const float*)d_in[2];
    const float* Wq  = (const float*)d_in[3];
    const float* bq  = (const float*)d_in[4];
    const float* Wk  = (const float*)d_in[5];
    const float* bk  = (const float*)d_in[6];
    const float* Wv  = (const float*)d_in[7];
    const float* bv  = (const float*)d_in[8];
    float* out = (float*)d_out;

    float* ws   = (float*)d_ws;
    float* kqs  = ws;            // 32*4*128       = 16384 floats
    float* cks  = ws + 16384;    // 128 floats (+pad to 256)
    float* vbar = ws + 16640;    // 65536*128      = 8388608 floats (~33.6 MB)

    ms_prep<<<BATCH * NSC, DIM, 0, stream>>>(Qin, Wq, bq, Wk, bk, kqs, cks);
    ms_attn<<<(BATCH * LOOK) / 8, 256, 0, stream>>>(Kin, Vin, kqs, cks, vbar);
    ms_proj<<<(BATCH * LOOK) / 128, 256, 0, stream>>>(vbar, Wv, bv, out);
}
